// LinearAttention_48060684042448
// MI455X (gfx1250) — matrix operand order repaired and checked by reading, not compiled, not run on hardware
//
#include <hip/hip_runtime.h>
#include <hip/hip_bf16.h>

// ---------------------------------------------------------------------------
// LinearAttention on MI455X (gfx1250), bf16 WMMA pipeline:
//   1) gemm<0,1>: Qp = elu(Xq^T Wq^T)+1        (bf16, token-major)
//   2) gemm<0,1>: Kp = elu(Xk^T Wk^T)+1
//   3) gemm<0,0>: Vp = Xv^T Wv^T
//   4) ksum:      Ksum[b][c] = sum_n Kp        (f32)
//   5) attn:      per token: S = Q K^T (1 wmma), Z = 1/(Q.Ksum+eps),
//                 out = (Z*S) V (2 K-padded wmma)
//   6) gemm<1,2>: d_out = Ot Wp^T + bp, written NCHW f32
// GEMM tile: 64(M) x 128(N), 8 waves, each wave 16x64 of C -> 4 wmma / K-step.
// ---------------------------------------------------------------------------

typedef __attribute__((ext_vector_type(8)))  float          v8f;
typedef __attribute__((ext_vector_type(16))) __bf16         v16bf;
typedef __attribute__((ext_vector_type(8)))  unsigned short v8u;

union Frag16 {
    v16bf b;
    v8u   h[2];
};

__device__ __forceinline__ v8f wmma_bf16(const Frag16& a, const Frag16& bb, v8f c) {
    return __builtin_amdgcn_wmma_f32_16x16x32_bf16(
        /*neg_a=*/false, a.b, /*neg_b=*/false, bb.b,
        /*c_mod=*/(short)0, c, /*reuse_a=*/false, /*reuse_b=*/false);
}

__device__ __forceinline__ float elu1f(float x) {
    // elu(x)+1 : x>0 -> x+1 ; else exp(x)
    return x > 0.0f ? x + 1.0f : __expf(x);
}

static constexpr int CDIM = 512;   // channels
static constexpr int NTOK = 4096;  // tokens per batch (64*64)
static constexpr int NB   = 8;     // batches
static constexpr int BM = 64, BN = 128, BK = 32;
static constexpr int LDA = 40;     // padded LDS row stride (elements, keeps 16B align)

// ---------------------------------------------------------------------------
// Tiled GEMM:  C[m, n] = sum_k A[m,k] * W[n,k]
//   ALAYOUT 0: A[m,k] = Af32[b][k][m]  (channel-major f32, NCHW tokens)
//   ALAYOUT 1: A[m,k] = Abf[b][m][k]   (token-major bf16)
//   EPI 0: store bf16 token-major; EPI 1: elu+1 then bf16; EPI 2: +bias, f32 NCHW
// ---------------------------------------------------------------------------
template <int ALAYOUT, int EPI>
__global__ __launch_bounds__(256)
void gemm_bf16_wmma(const float* __restrict__ Af32,
                    const __bf16* __restrict__ Abf,
                    const float* __restrict__ W,
                    const float* __restrict__ bias,
                    __bf16* __restrict__ outBf,
                    float* __restrict__ outF32)
{
    __shared__ __align__(16) __bf16 As[BM * LDA];   //  5 KB
    __shared__ __align__(16) __bf16 Bs[BN * LDA];   // 10 KB

    const int bm   = blockIdx.x * BM;
    const int bn   = blockIdx.y * BN;
    const int b    = blockIdx.z;
    const int tid  = threadIdx.x;
    const int lane = tid & 31;
    const int wave = tid >> 5;
    const int wm   = wave & 3;   // 4 waves tile M (16 rows each)
    const int wn   = wave >> 2;  // 2 waves tile N (64 cols each)

    v8f acc[4] = {v8f{}, v8f{}, v8f{}, v8f{}};

    for (int kb = 0; kb < CDIM; kb += BK) {
        // ---- stage A tile -> As[m][k] (bf16) ----
        if constexpr (ALAYOUT == 0) {
            const int mloc = tid & 63;
            const int k0   = tid >> 6;  // 0..3
            #pragma unroll
            for (int i = 0; i < 8; ++i) {
                const int k = k0 + i * 4;
                float x = Af32[((size_t)b * CDIM + kb + k) * NTOK + bm + mloc];
                As[mloc * LDA + k] = (__bf16)x;
            }
        } else {
            const int kloc = tid & 31;
            const int m0   = tid >> 5;  // 0..7
            #pragma unroll
            for (int i = 0; i < 8; ++i) {
                const int m = m0 + i * 8;
                As[m * LDA + kloc] =
                    Abf[((size_t)b * NTOK + bm + m) * CDIM + kb + kloc];
            }
        }
        // ---- stage B tile -> Bs[n][k] (bf16) straight from W[n][k] ----
        {
            const int kloc = tid & 31;
            const int n0   = tid >> 5;
            #pragma unroll
            for (int i = 0; i < 16; ++i) {
                const int n = n0 + i * 8;
                float x = W[(size_t)(bn + n) * CDIM + kb + kloc];
                Bs[n * LDA + kloc] = (__bf16)x;
            }
            if (kb + BK < CDIM)
                __builtin_prefetch(&W[(size_t)(bn + n0) * CDIM + kb + BK], 0, 3);
        }
        __syncthreads();

        // ---- fragments per ISA layout ----
        Frag16 fa, fb[4];
        {
            const int row   = wm * 16 + (lane & 15);
            const int kbase = (lane < 16) ? 0 : 8;
            const __bf16* p = &As[row * LDA + kbase];
            fa.h[0] = *(const v8u*)p;          // K kbase..kbase+7
            fa.h[1] = *(const v8u*)(p + 16);   // K kbase+16..kbase+23
        }
        {
            const int kbase = (lane < 16) ? 0 : 8;
            const int nrow  = wn * 64 + (lane & 15);
            #pragma unroll
            for (int j = 0; j < 4; ++j) {
                const __bf16* p = &Bs[(nrow + j * 16) * LDA + kbase];
                fb[j].h[0] = *(const v8u*)p;
                fb[j].h[1] = *(const v8u*)(p + 16);
            }
        }
        #pragma unroll
        for (int j = 0; j < 4; ++j) acc[j] = wmma_bf16(fa, fb[j], acc[j]);
        __syncthreads();
    }

    // ---- epilogue ----
    const int rowBase = bm + wm * 16 + ((lane >> 4) << 3);
    #pragma unroll
    for (int j = 0; j < 4; ++j) {
        const int col = bn + wn * 64 + j * 16 + (lane & 15);
        #pragma unroll
        for (int r = 0; r < 8; ++r) {
            const int m = rowBase + r;
            float v0 = acc[j][r];
            if constexpr (EPI == 1) { v0 = elu1f(v0); }
            if constexpr (EPI == 2) {
                outF32[((size_t)b * CDIM + col) * NTOK + m] = v0 + bias[col];
            } else {
                outBf[((size_t)b * NTOK + m) * CDIM + col] = (__bf16)v0;
            }
        }
    }
}

// ---------------------------------------------------------------------------
// Ksum[b][c] = sum_n Kp[b][n][c]   (4096 threads, coalesced over c)
// ---------------------------------------------------------------------------
__global__ __launch_bounds__(256)
void ksum_kernel(const __bf16* __restrict__ Kp, float* __restrict__ Ksum)
{
    const int idx = blockIdx.x * 256 + threadIdx.x;  // 0..4095
    const int b = idx >> 9;
    const int c = idx & 511;
    const __bf16* p = Kp + (size_t)b * NTOK * CDIM + c;
    float s = 0.0f;
    for (int n = 0; n < NTOK; ++n) s += (float)p[(size_t)n * CDIM];
    Ksum[idx] = s;
}

// ---------------------------------------------------------------------------
// Per-token attention: 1 wave per token (8 tokens / 256-thread block).
//   S  = Q K^T            (16x16, K=32)       -> 1 wmma
//   Z  = 1/(Q . Ksum + e) (per head)
//   out = (Z*S) V         (16x32, K=16 pad32) -> 2 wmma
// ---------------------------------------------------------------------------
__global__ __launch_bounds__(256)
void attn_kernel(const __bf16* __restrict__ Qp, const __bf16* __restrict__ Kp,
                 const __bf16* __restrict__ Vp, const float* __restrict__ Ksum,
                 __bf16* __restrict__ Ot)
{
    __shared__ __align__(16) __bf16 STl[8][16 * 16];  // Z-scaled S, [h][h']
    __shared__ __align__(16) __bf16 VTl[8][32 * 16];  // V^T, [m][h']
    __shared__ float                Zl[8][16];

    const int wave = threadIdx.x >> 5;
    const int lane = threadIdx.x & 31;
    const size_t tok = (size_t)blockIdx.x * 8 + wave;
    const int b = (int)(tok >> 12);
    const size_t base = tok * CDIM;

    __bf16* ST = STl[wave];
    __bf16* VT = VTl[wave];
    float*  Z  = Zl[wave];

    // ---- S = Q K^T : both fragments contiguous per lane in token-major bf16 ----
    Frag16 fq, fk;
    {
        const int h  = lane & 15;
        const int kb = (lane < 16) ? 0 : 8;
        const __bf16* p = Qp + base + h * 32 + kb;
        fq.h[0] = *(const v8u*)p;
        fq.h[1] = *(const v8u*)(p + 16);
    }
    {
        const int hp = lane & 15;
        const int kb = (lane < 16) ? 0 : 8;
        const __bf16* p = Kp + base + hp * 32 + kb;
        fk.h[0] = *(const v8u*)p;
        fk.h[1] = *(const v8u*)(p + 16);
    }
    v8f s = {};
    s = wmma_bf16(fq, fk, s);

    // ---- Z[h] = 1/(sum_d Q[h,d]*Ksum[b,h,d] + eps) ----
    {
        const int h  = lane & 15;
        const int dh = (lane >> 4) * 16;
        const __bf16* qr = Qp + base + h * 32 + dh;
        const float*  kr = Ksum + (size_t)b * CDIM + h * 32 + dh;
        float part = 0.0f;
        #pragma unroll
        for (int d = 0; d < 16; ++d) part += (float)qr[d] * kr[d];
        part += __shfl_xor(part, 16, 32);
        if (lane < 16) Z[h] = 1.0f / (part + 1e-5f);
    }

    // ---- stage V^T into LDS: VT[m][h'] = V[h'][m] ----
    {
        const int hp = lane & 15;
        const int mh = (lane >> 4) * 16;
        const __bf16* v = Vp + base + hp * 32 + mh;
        #pragma unroll
        for (int j = 0; j < 16; ++j) VT[(mh + j) * 16 + hp] = v[j];
    }
    __syncthreads();

    // ---- stage Z-scaled S: ST[h][h'] = Z[h]*S[h][h'] ----
    {
        const int col   = lane & 15;
        const int rbase = (lane >> 4) * 8;
        #pragma unroll
        for (int r = 0; r < 8; ++r) {
            const int row = rbase + r;
            ST[row * 16 + col] = (__bf16)(s[r] * Z[row]);
        }
    }
    __syncthreads();

    // ---- out = S' V : A = S' padded K16->32, B = V^T cols, hi-K zero ----
    const v8u zero8 = {0, 0, 0, 0, 0, 0, 0, 0};
    Frag16 fa2;
    {
        const int row = lane & 15;
        const int kb  = (lane < 16) ? 0 : 8;
        fa2.h[0] = *(const v8u*)(ST + row * 16 + kb);
        fa2.h[1] = zero8;  // K=16..31 padding
    }
    Frag16 fb0, fb1;
    {
        const int kb = (lane < 16) ? 0 : 8;
        const __bf16* p0 = VT + (lane & 15) * 16 + kb;        // m = 0..15
        fb0.h[0] = *(const v8u*)p0;
        fb0.h[1] = zero8;  // K=16..31 padding
        const __bf16* p1 = VT + (16 + (lane & 15)) * 16 + kb; // m = 16..31
        fb1.h[0] = *(const v8u*)p1;
        fb1.h[1] = zero8;  // K=16..31 padding
    }
    v8f o0 = {}, o1 = {};
    o0 = wmma_bf16(fa2, fb0, o0);
    o1 = wmma_bf16(fa2, fb1, o1);

    // ---- store out token: channel = h*32 + m ----
    {
        const int col   = lane & 15;
        const int rbase = (lane >> 4) * 8;
        #pragma unroll
        for (int r = 0; r < 8; ++r) {
            const int h = rbase + r;
            Ot[base + h * 32 + col]      = (__bf16)o0[r];
            Ot[base + h * 32 + 16 + col] = (__bf16)o1[r];
        }
    }
}

// ---------------------------------------------------------------------------
extern "C" void kernel_launch(void* const* d_in, const int* in_sizes, int n_in,
                              void* d_out, int out_size, void* d_ws, size_t ws_size,
                              hipStream_t stream)
{
    (void)in_sizes; (void)n_in; (void)out_size; (void)ws_size;

    const float* q  = (const float*)d_in[0];
    const float* k  = (const float*)d_in[1];
    const float* v  = (const float*)d_in[2];
    const float* Wq = (const float*)d_in[3];
    const float* Wk = (const float*)d_in[4];
    const float* Wv = (const float*)d_in[5];
    const float* Wp = (const float*)d_in[6];
    const float* bp = (const float*)d_in[7];
    float* out = (float*)d_out;

    const size_t tokBytes = (size_t)NB * NTOK * CDIM * sizeof(__bf16);  // 32 MB
    char* ws = (char*)d_ws;
    __bf16* Qp   = (__bf16*)(ws + 0 * tokBytes);
    __bf16* Kp   = (__bf16*)(ws + 1 * tokBytes);
    __bf16* Vp   = (__bf16*)(ws + 2 * tokBytes);
    __bf16* Ot   = (__bf16*)(ws + 3 * tokBytes);
    float*  Ksum = (float*)(ws + 4 * tokBytes);

    dim3 grid(NTOK / BM, CDIM / BN, NB);  // (64, 4, 8)
    dim3 block(256);

    gemm_bf16_wmma<0, 1><<<grid, block, 0, stream>>>(q, nullptr, Wq, nullptr, Qp, nullptr);
    gemm_bf16_wmma<0, 1><<<grid, block, 0, stream>>>(k, nullptr, Wk, nullptr, Kp, nullptr);
    gemm_bf16_wmma<0, 0><<<grid, block, 0, stream>>>(v, nullptr, Wv, nullptr, Vp, nullptr);

    ksum_kernel<<<dim3(NB * CDIM / 256), block, 0, stream>>>(Kp, Ksum);

    attn_kernel<<<dim3(NB * NTOK / 8), block, 0, stream>>>(Qp, Kp, Vp, Ksum, Ot);

    gemm_bf16_wmma<1, 2><<<grid, block, 0, stream>>>(nullptr, Ot, Wp, bp, nullptr, out);
}
